// IIRFilter_4148938408287
// MI455X (gfx1250) — matrix operand order repaired and checked by reading, not compiled, not run on hardware
//
#include <hip/hip_runtime.h>
#include <stdint.h>

typedef __attribute__((ext_vector_type(16))) _Float16 v16h;
typedef __attribute__((ext_vector_type(8)))  float    v8f;
typedef _Float16 half_t;

#define T_LEN   131072
#define NCH     64
#define S_LEN   (T_LEN + 512)      // padded per-channel stride in halfs (48 pre, 464 post)
#define PAD_LO  48
#define NTAPS   48                 // tail |pole|^48 ~ 1e-9 rel << f16-split error ~2e-7
#define KB      2                  // contraction window = NTAPS + 16 = 64 = 2 x K32
#define TILE_T  256                // outputs per wave-tile (16x16)
#define NTILES  (T_LEN / TILE_T)   // 512 tiles per channel
#define TPW     8                  // tiles per wave (512 % 8 == 0: no channel crossing)

// workspace layout (units: halfs):
//   [0 .. 2047]          Bconst tiles: (kb, split, lane) x 16 halfs, 32B/lane
//   [8192 ..)            hiArr: NCH * S_LEN   (padded f16-hi of x)
//   [8192+NCH*S_LEN ..)  loArr: NCH * S_LEN   (padded f16-lo residual of x)
#define BC_OFF 0
#define HI_OFF 8192
#define LO_OFF (HI_OFF + (long)NCH * S_LEN)

// ---------------------------------------------------------------------------
// Kernel 1: impulse response h[0..47] from (b,a); emit banded Toeplitz H^T
// tiles in WMMA B-matrix layout, both f16 splits.
// B layout (16-bit B, 32x16): lanes 0-15 carry N=lane with K={0..7,16..23} packed
// 2-per-VGPR; lanes 16-31 carry N=lane-16 with K={8..15,24..31}.
// B[k=u_local, n=m] = h[m + 48 - (32*kb + u_local)]   (0 outside [0,47])
// ---------------------------------------------------------------------------
__global__ void iir_setup_taps(const float* __restrict__ b,
                               const float* __restrict__ a,
                               half_t* __restrict__ wsh) {
    __shared__ float hsh[NTAPS];
    const int lane = threadIdx.x;
    if (lane == 0) {
        float a0 = a[0];
        float b0 = b[0] / a0, b1 = b[1] / a0, b2 = b[2] / a0;
        float a1 = a[1] / a0, a2 = a[2] / a0;
        float s0 = 0.f, s1 = 0.f;
        for (int j = 0; j < NTAPS; ++j) {
            float xin = (j == 0) ? 1.f : 0.f;
            float yv = b0 * xin + s0;
            s0 = b1 * xin - a1 * yv + s1;
            s1 = b2 * xin - a2 * yv;
            hsh[j] = yv;
        }
    }
    __syncthreads();
    const int m  = lane & 15;          // N column (time-in-block)
    const int hh = (lane >> 4) * 16;   // K half offset
    for (int kb = 0; kb < KB; ++kb) {
        for (int split = 0; split < 2; ++split) {
            half_t* dst = wsh + BC_OFF + (((kb * 2 + split) * 32) + lane) * 16;
            for (int e = 0; e < 16; ++e) {
                int u = kb * 32 + (hh >> 1) + e + ((e >> 3) << 3);
                int j = m + PAD_LO - u;
                float v = (j >= 0 && j < NTAPS) ? hsh[j] : 0.f;
                half_t vh = (half_t)v;
                dst[e] = (split == 0) ? vh : (half_t)(v - (float)vh);
            }
        }
    }
}

// ---------------------------------------------------------------------------
// Kernel 2: split x into padded per-channel f16 hi/lo arrays.
// Position p in [0, S_LEN): value = x[ch, p-48] inside the channel, else 0.
// The 48 leading zeros realize the filter's zero initial state exactly.
// ---------------------------------------------------------------------------
__global__ void iir_split_x(const float* __restrict__ x,
                            half_t* __restrict__ wsh) {
    const int ch = blockIdx.y;
    const long p = ((long)blockIdx.x * blockDim.x + threadIdx.x) * 2;
    if (p >= S_LEN) return;
    const long basec = (long)ch * S_LEN;
    const long xb    = (long)ch * T_LEN;
    float v0 = 0.f, v1 = 0.f;
    long t = p - PAD_LO;
    if (t >= 0 && t < T_LEN)         v0 = x[xb + t];
    if (t + 1 >= 0 && t + 1 < T_LEN) v1 = x[xb + t + 1];
    half_t h0 = (half_t)v0, h1 = (half_t)v1;
    half_t l0 = (half_t)(v0 - (float)h0), l1 = (half_t)(v1 - (float)h1);
    union { half_t h[2]; uint32_t u; } ph, pl;
    ph.h[0] = h0; ph.h[1] = h1;
    pl.h[0] = l0; pl.h[1] = l1;
    *(uint32_t*)(wsh + HI_OFF + basec + p) = ph.u;
    *(uint32_t*)(wsh + LO_OFF + basec + p) = pl.u;
}

// ---------------------------------------------------------------------------
// Kernel 3: FIR-as-WMMA with software double-buffering. Each wave computes
// TPW consecutive 16x16 tiles of Y^T = sum_kb W^T(kb) * H^T(kb); 3 split
// products (hi*hi + hi*lo + lo*hi), f32 accumulate. B (taps) tiles live in
// VGPRs; tile it+1's eight b128 A-loads are issued before tile it's WMMA
// chain so load latency overlaps matrix work.
// A (16-bit, 16x32) layout: lane l<16 -> M=l, K = {0..7,16..23}; lanes 16-31
// -> K = {8..15,24..31}; each half = two aligned b128 runs.
// ---------------------------------------------------------------------------
union AU { v16h v; uint4 q[2]; };

__global__ __launch_bounds__(256) void iir_fir_wmma(const half_t* __restrict__ wsh,
                                                    float* __restrict__ y) {
    const int lane   = threadIdx.x & 31;
    const int waveId = threadIdx.x >> 5;
    const int wave   = blockIdx.x * 8 + waveId;      // 8 waves / 256-thread block
    const int tile0  = wave * TPW;                   // first tile of this wave
    const int ch     = tile0 >> 9;                   // NTILES = 512
    const int tIdx0  = tile0 & (NTILES - 1);
    const int lane15 = lane & 15;
    const int kA     = (lane >> 4) * 8;

    // Hold the 4 constant B tiles (H^T hi/lo for kb=0,1) in registers.
    const v16h bh0 = *(const v16h*)(wsh + BC_OFF + ((0 * 32) + lane) * 16);
    const v16h bl0 = *(const v16h*)(wsh + BC_OFF + ((1 * 32) + lane) * 16);
    const v16h bh1 = *(const v16h*)(wsh + BC_OFF + ((2 * 32) + lane) * 16);
    const v16h bl1 = *(const v16h*)(wsh + BC_OFF + ((3 * 32) + lane) * 16);

    // x-time of A element (n, u) is t0 + 16n + u - 48; array has +48 pre-pad,
    // so half-index = ch*S + t0 + 16n + u,  u = 32*kb + kA + {0..7, 16..23}.
    const long aBase0 = (long)ch * S_LEN + (long)tIdx0 * TILE_T + 16 * lane15 + kA;
    long outBase = (long)ch * T_LEN + (long)tIdx0 * TILE_T
                 + 128 * (lane >> 4) + lane15;
    const half_t* hiA = wsh + HI_OFF;
    const half_t* loA = wsh + LO_OFF;

    AU buf[2][4];   // double buffer: [parity][hi0, lo0, hi1, lo1]

    auto loadA = [&](long base, AU* dst) {
        dst[0].q[0] = *(const uint4*)(hiA + base);
        dst[0].q[1] = *(const uint4*)(hiA + base + 16);
        dst[1].q[0] = *(const uint4*)(loA + base);
        dst[1].q[1] = *(const uint4*)(loA + base + 16);
        dst[2].q[0] = *(const uint4*)(hiA + base + 32);
        dst[2].q[1] = *(const uint4*)(hiA + base + 48);
        dst[3].q[0] = *(const uint4*)(loA + base + 32);
        dst[3].q[1] = *(const uint4*)(loA + base + 48);
    };

    loadA(aBase0, buf[0]);

    #pragma unroll
    for (int it = 0; it < TPW; ++it) {
        const AU* c = buf[it & 1];
        if (it + 1 < TPW)   // prefetch next tile while this tile's WMMAs run
            loadA(aBase0 + (long)(it + 1) * TILE_T, buf[(it + 1) & 1]);

        v8f acc = {};
        acc = __builtin_amdgcn_wmma_f32_16x16x32_f16(false, c[0].v, false, bh0,
                                                     (short)0, acc, false, false);
        acc = __builtin_amdgcn_wmma_f32_16x16x32_f16(false, c[0].v, false, bl0,
                                                     (short)0, acc, false, false);
        acc = __builtin_amdgcn_wmma_f32_16x16x32_f16(false, c[1].v, false, bh0,
                                                     (short)0, acc, false, false);
        acc = __builtin_amdgcn_wmma_f32_16x16x32_f16(false, c[2].v, false, bh1,
                                                     (short)0, acc, false, false);
        acc = __builtin_amdgcn_wmma_f32_16x16x32_f16(false, c[2].v, false, bl1,
                                                     (short)0, acc, false, false);
        acc = __builtin_amdgcn_wmma_f32_16x16x32_f16(false, c[3].v, false, bh1,
                                                     (short)0, acc, false, false);

        // D layout: VGPR r, lane l -> row M = r + 8*(l>=16) = n-tile, col = l&15.
        // Each store covers two coalesced 64B lines.
        #pragma unroll
        for (int r = 0; r < 8; ++r) {
            y[outBase + 16 * r] = acc[r];
        }
        outBase += TILE_T;
    }
}

// ---------------------------------------------------------------------------
extern "C" void kernel_launch(void* const* d_in, const int* in_sizes, int n_in,
                              void* d_out, int out_size, void* d_ws, size_t ws_size,
                              hipStream_t stream) {
    (void)in_sizes; (void)n_in; (void)out_size; (void)ws_size;
    const float* x = (const float*)d_in[0];
    const float* b = (const float*)d_in[1];
    const float* a = (const float*)d_in[2];
    float* yout = (float*)d_out;
    half_t* wsh = (half_t*)d_ws;

    iir_setup_taps<<<1, 32, 0, stream>>>(b, a, wsh);

    dim3 gConv(S_LEN / 512, NCH);   // 257 x 64 blocks, 2 elems/thread
    iir_split_x<<<gConv, 256, 0, stream>>>(x, wsh);

    const int totalWaves = (NCH * NTILES) / TPW;   // 4096 waves
    iir_fir_wmma<<<totalWaves / 8, 256, 0, stream>>>(wsh, yout);
}
